// HSGNN_47184510714070
// MI455X (gfx1250) — compile-verified, matrix operand order checked by reading, not run on hardware
//
#include <hip/hip_runtime.h>

typedef __attribute__((ext_vector_type(16))) _Float16 v16h;
typedef __attribute__((ext_vector_type(8)))  float    v8f;

#define DI __device__ __forceinline__

DI float lrelu_(float v, float s) { return v < 0.f ? s * v : v; }

DI float atomicMaxF(float* a, float v) {
  // int-ordering trick: works for mixed signs with -inf init (0xFF800000)
  return (v >= 0.f)
    ? __int_as_float(atomicMax((int*)a, __float_as_int(v)))
    : __uint_as_float(atomicMin((unsigned int*)a, __float_as_uint(v)));
}

// ---------------------------------------------------------------- fill / pad
__global__ void k_fill(float* p, float v, long n) {
  long i = (long)blockIdx.x * blockDim.x + threadIdx.x;
  if (i < n) p[i] = v;
}

__global__ void k_padcopy(const float* __restrict__ src, float* __restrict__ dst,
                          long rows, int Kreal, int Kpad) {
  long t = (long)blockIdx.x * blockDim.x + threadIdx.x;
  if (t >= rows * Kpad) return;
  long r = t / Kpad; int c = (int)(t - r * Kpad);
  dst[t] = (c < Kreal) ? src[r * Kreal + c] : 0.f;
}

// ---------------------------------------------------------------- B prep
// Swizzle weight B[K,N] (f32, row-major) into fragment-ready f16:
// Bp[((kt*NT + nt)*32 + lane)*16 + e], zero-padded to Kpad (mult of 32).
__global__ void k_prepB(const float* __restrict__ B, _Float16* __restrict__ Bp,
                        int Kreal, int Kpad, int N) {
  int t = blockIdx.x * blockDim.x + threadIdx.x;
  int total = (Kpad / 32) * (N / 16) * 32;
  if (t >= total) return;
  int lane = t & 31, tile = t >> 5;
  int NT = N / 16;
  int nt = tile % NT, kt = tile / NT;
  int lo = lane & 15, hi = lane >> 4;
  int col = nt * 16 + lo;
  _Float16* out = Bp + (long)t * 16;
#pragma unroll
  for (int e = 0; e < 16; ++e) {
    int k = kt * 32 + e + ((e >= 8) ? 8 : 0) + hi * 8;   // CDNA5 16-bit frag K-order
    out[e] = (_Float16)((k < Kreal) ? B[(long)k * N + col] : 0.f);
  }
}

// ---------------------------------------------------------------- WMMA GEMM
// One wave computes a 16 x (NT*16) strip. A:[M,K] f32 (K mult of 32),
// Bp: fragment-ready f16. 8 waves / block.
template <int NT>
__global__ __launch_bounds__(256) void k_gemm(
    const float* __restrict__ A, const _Float16* __restrict__ Bp,
    const float* __restrict__ bias, float* __restrict__ C,
    int M, int K, int ldc, int col0, int act) {
  int lane = threadIdx.x & 31;
  int wid  = threadIdx.x >> 5;
  int lo = lane & 15, hi = lane >> 4;
  int tM = blockIdx.x * 8 + wid;
  int row = tM * 16 + lo;
  if (row >= M) row = M - 1;          // clamp: extra rows computed, never stored
  const int K32 = K >> 5;
  v8f acc[NT] = {};
  for (int kt = 0; kt < K32; ++kt) {
    const float* arow = A + (long)row * K + kt * 32 + hi * 8;
    float4 a0 = *(const float4*)(arow);
    float4 a1 = *(const float4*)(arow + 4);
    float4 a2 = *(const float4*)(arow + 16);
    float4 a3 = *(const float4*)(arow + 20);
    v16h af;
    af[0]  = (_Float16)a0.x; af[1]  = (_Float16)a0.y;
    af[2]  = (_Float16)a0.z; af[3]  = (_Float16)a0.w;
    af[4]  = (_Float16)a1.x; af[5]  = (_Float16)a1.y;
    af[6]  = (_Float16)a1.z; af[7]  = (_Float16)a1.w;
    af[8]  = (_Float16)a2.x; af[9]  = (_Float16)a2.y;
    af[10] = (_Float16)a2.z; af[11] = (_Float16)a2.w;
    af[12] = (_Float16)a3.x; af[13] = (_Float16)a3.y;
    af[14] = (_Float16)a3.z; af[15] = (_Float16)a3.w;
#pragma unroll
    for (int nt = 0; nt < NT; ++nt) {
      v16h bf = *(const v16h*)(Bp + ((long)(kt * NT + nt) * 32 + lane) * 16);
      acc[nt] = __builtin_amdgcn_wmma_f32_16x16x32_f16(
          false, af, false, bf, (short)0, acc[nt], false, false);
    }
  }
#pragma unroll
  for (int nt = 0; nt < NT; ++nt) {
#pragma unroll
    for (int r = 0; r < 8; ++r) {
      int m = tM * 16 + r + hi * 8;                      // C/D layout
      if (m < M) {
        float v = acc[nt][r] + (bias ? bias[nt * 16 + lo] : 0.f);
        if (act == 1) v = fmaxf(v, 0.f);
        else if (act == 2) v = lrelu_(v, 0.01f);
        C[(long)m * ldc + col0 + nt * 16 + lo] = v;
      }
    }
  }
}

// ---------------------------------------------------------------- GAT pieces
__global__ void k_attn(const float* __restrict__ z, const float* __restrict__ al,
                       const float* __restrict__ ar, float* el, float* er, long n) {
  long t = (long)blockIdx.x * blockDim.x + threadIdx.x;
  if (t >= n) return;
  long node = t >> 2; int h = t & 3;
  const float4* zp = (const float4*)(z + node * 128 + h * 32);
  const float4* ap = (const float4*)(al + h * 32);
  const float4* bp = (const float4*)(ar + h * 32);
  float a = 0.f, b = 0.f;
#pragma unroll
  for (int q = 0; q < 8; ++q) {
    float4 zv = zp[q], av = ap[q], bv = bp[q];
    a += zv.x * av.x + zv.y * av.y + zv.z * av.z + zv.w * av.w;
    b += zv.x * bv.x + zv.y * bv.y + zv.z * bv.z + zv.w * bv.w;
  }
  el[t] = a; er[t] = b;
}

__global__ void k_edge_max(const int* __restrict__ src, const int* __restrict__ dst,
                           const float* __restrict__ el, const float* __restrict__ er,
                           float* mx, long n) {
  long t = (long)blockIdx.x * blockDim.x + threadIdx.x;
  if (t >= n) return;
  long e = t >> 2; int h = t & 3;
  long s = src[e], d = dst[e];
  float v = lrelu_(el[s * 4 + h] + er[d * 4 + h], 0.2f);
  atomicMaxF(&mx[d * 4 + h], v);
}

__global__ void k_edge_soft(const int* __restrict__ src, const int* __restrict__ dst,
                            const float* __restrict__ el, const float* __restrict__ er,
                            const float* __restrict__ mx, const float* __restrict__ z,
                            float* denom, float* num, long n) {
  long t = (long)blockIdx.x * blockDim.x + threadIdx.x;
  if (t >= n) return;
  long e = t >> 2; int h = t & 3;
  long s = src[e], d = dst[e];
  float v = lrelu_(el[s * 4 + h] + er[d * 4 + h], 0.2f);
  float p = expf(v - mx[d * 4 + h]);
  atomicAdd(&denom[d * 4 + h], p);
  const float4* zp = (const float4*)(z + s * 128 + h * 32);
  float* np = num + d * 128 + h * 32;
#pragma unroll
  for (int q = 0; q < 8; ++q) {
    float4 zv = zp[q];
    atomicAdd(&np[q * 4 + 0], p * zv.x);
    atomicAdd(&np[q * 4 + 1], p * zv.y);
    atomicAdd(&np[q * 4 + 2], p * zv.z);
    atomicAdd(&np[q * 4 + 3], p * zv.w);
  }
}

__global__ void k_gat_fin(const float* __restrict__ dn, const float* __restrict__ b,
                          float* num, long n) {  // n = N*128, in-place on num
  long t = (long)blockIdx.x * blockDim.x + threadIdx.x;
  if (t >= n) return;
  long node = t >> 7; int f = t & 127; int h = f >> 5;
  float v = num[t] / fmaxf(dn[node * 4 + h], 1e-9f) + b[f];
  num[t] = fmaxf(v, 0.f);
}

// ---------------------------------------------------------------- scatter ops
__global__ void k_scat_add4(const int* __restrict__ src, const int* __restrict__ dst,
                            const float* __restrict__ x, float* s, long n, int F4) {
  long t = (long)blockIdx.x * blockDim.x + threadIdx.x;   // n = E * F4
  if (t >= n) return;
  long e = t / F4; int q = (int)(t - e * F4);
  float4 v = *(const float4*)(x + ((long)src[e] * F4 + q) * 4);
  float* sp = s + ((long)dst[e] * F4 + q) * 4;
  atomicAdd(&sp[0], v.x); atomicAdd(&sp[1], v.y);
  atomicAdd(&sp[2], v.z); atomicAdd(&sp[3], v.w);
}

__global__ void k_scat_max4(const int* __restrict__ src, const int* __restrict__ dst,
                            const float* __restrict__ x, float* m, long n, int F4) {
  long t = (long)blockIdx.x * blockDim.x + threadIdx.x;
  if (t >= n) return;
  long e = t / F4; int q = (int)(t - e * F4);
  float4 v = *(const float4*)(x + ((long)src[e] * F4 + q) * 4);
  float* mp = m + ((long)dst[e] * F4 + q) * 4;
  atomicMaxF(&mp[0], v.x); atomicMaxF(&mp[1], v.y);
  atomicMaxF(&mp[2], v.z); atomicMaxF(&mp[3], v.w);
}

__global__ void k_deg(const int* __restrict__ dst, float* deg, long E) {
  long t = (long)blockIdx.x * blockDim.x + threadIdx.x;
  if (t < E) atomicAdd(&deg[dst[t]], 1.f);
}

__global__ void k_mean_div(float* s, const float* __restrict__ deg, long n, int F) {
  long t = (long)blockIdx.x * blockDim.x + threadIdx.x;
  if (t >= n) return;
  s[t] /= fmaxf(deg[t / F], 1.f);
}

__global__ void k_gin_pre(const float* __restrict__ h, const float* __restrict__ agg,
                          const float* __restrict__ epsp, float* X, long n) {
  long t = (long)blockIdx.x * blockDim.x + threadIdx.x;
  if (t >= n) return;
  X[t] = (1.f + epsp[0]) * h[t] + agg[t];
}

__global__ void k_fix(float* m, long n) {  // -inf (no in-edges) -> 0
  long t = (long)blockIdx.x * blockDim.x + threadIdx.x;
  if (t >= n) return;
  float v = m[t];
  if (v < -1e30f) m[t] = 0.f;
}

__global__ void k_concat(const float* __restrict__ pooled, const float* __restrict__ pimg,
                         float* out, long n) {  // n = NC2*356
  long t = (long)blockIdx.x * blockDim.x + threadIdx.x;
  if (t >= n) return;
  long node = t / 356; int f = (int)(t - node * 356);
  out[t] = (f < 256) ? pooled[node * 256 + f] : pimg[node * 100 + (f - 256)];
}

// ---------------------------------------------------------------- BatchNorm
__global__ void k_bn_stats(const float* __restrict__ z, float* stat, int n, int F) {
  int f = threadIdx.x;
  if (f >= F) return;
  float s = 0.f, s2 = 0.f;
  for (int i = 0; i < n; ++i) { float v = z[(long)i * F + f]; s += v; s2 += v * v; }
  float mu = s / n;
  stat[f] = mu;
  stat[F + f] = s2 / n - mu * mu;  // biased variance
}

__global__ void k_bn_apply(float* z, const float* __restrict__ stat,
                           const float* __restrict__ gamma, const float* __restrict__ beta,
                           long n, int F) {
  long t = (long)blockIdx.x * blockDim.x + threadIdx.x;
  if (t >= n) return;
  int f = (int)(t % F);
  z[t] = (z[t] - stat[f]) * rsqrtf(stat[F + f] + 1e-5f) * gamma[f] + beta[f];
}

// ---------------------------------------------------------------- readout
__global__ void k_seg_sum4(const int* __restrict__ seg, const float* __restrict__ x,
                           float* r, long n, int F4, int ldr, int col0) {
  long t = (long)blockIdx.x * blockDim.x + threadIdx.x;   // n = rows * F4
  if (t >= n) return;
  long i = t / F4; int q = (int)(t - i * F4);
  float4 v = *(const float4*)(x + (i * F4 + q) * 4);
  float* rp = r + (long)seg[i] * ldr + col0 + q * 4;
  atomicAdd(&rp[0], v.x); atomicAdd(&rp[1], v.y);
  atomicAdd(&rp[2], v.z); atomicAdd(&rp[3], v.w);
}

__global__ void k_seg_cnt(const int* __restrict__ seg, float* cnt, long n) {
  long t = (long)blockIdx.x * blockDim.x + threadIdx.x;
  if (t < n) atomicAdd(&cnt[seg[t]], 1.f);
}

__global__ void k_rdiv(float* r, const float* __restrict__ cA, const float* __restrict__ c2,
                       const float* __restrict__ c3, long n) {  // n = 64*512
  long t = (long)blockIdx.x * blockDim.x + threadIdx.x;
  if (t >= n) return;
  long b = t >> 9; int c = (int)(t & 511);
  float cn = (c < 256) ? cA[b] : (c < 384 ? c2[b] : c3[b]);
  r[t] /= fmaxf(cn, 1.f);
}

// ================================================================ host
static inline long cdivl(long a, long b) { return (a + b - 1) / b; }

extern "C" void kernel_launch(void* const* d_in, const int* in_sizes, int n_in,
                              void* d_out, int out_size, void* d_ws, size_t ws_size,
                              hipStream_t stream) {
  (void)in_sizes; (void)n_in; (void)out_size; (void)ws_size;
  constexpr long NA = 200000, EA = 1000000, NC2 = 20000, NC3 = 2000;
  constexpr long EG1 = 200000, EI2 = 200000, EG2 = 20000, EI3 = 20000;

  const float* feats = (const float*)d_in[0];
  const float* pimg  = (const float*)d_in[1];

  // params (pytree flatten: dict keys sorted; lists in order)
  struct GatP { const float *W, *al, *ar, *b; };
  struct GinP { const float *W, *b, *eps; };
  GatP gat[2][2]; GinP gin[2][2];
  int pi = 2;
  for (int t = 0; t < 2; ++t) {
    for (int l = 0; l < 2; ++l) {
      gat[t][l] = { (const float*)d_in[pi], (const float*)d_in[pi+1],
                    (const float*)d_in[pi+2], (const float*)d_in[pi+3] };
      pi += 4;
    }
    for (int l = 0; l < 2; ++l) {
      gin[t][l] = { (const float*)d_in[pi], (const float*)d_in[pi+1],
                    (const float*)d_in[pi+2] };
      pi += 3;
    }
  }
  const float* h2W[2]; const float* h2eps[2];
  for (int l = 0; l < 2; ++l) { h2W[l] = (const float*)d_in[pi]; h2eps[l] = (const float*)d_in[pi+1]; pi += 2; }
  const float* h3W0 = (const float*)d_in[pi++];
  const float* h3W1 = (const float*)d_in[pi++];
  const float* h3beta = (const float*)d_in[pi++];
  const float* h3eps = (const float*)d_in[pi++];
  const float* h3gamma = (const float*)d_in[pi++];
  const float* oW0 = (const float*)d_in[pi++];
  const float* oW1 = (const float*)d_in[pi++];

  const int* src_b[2] = { (const int*)d_in[pi], (const int*)d_in[pi+2] };
  const int* dst_b[2] = { (const int*)d_in[pi+1], (const int*)d_in[pi+3] };
  pi += 4;
  const int* g1_src = (const int*)d_in[pi++]; const int* g1_dst = (const int*)d_in[pi++];
  const int* i2_src = (const int*)d_in[pi++]; const int* i2_dst = (const int*)d_in[pi++];
  const int* g2_src = (const int*)d_in[pi++]; const int* g2_dst = (const int*)d_in[pi++];
  const int* i3_src = (const int*)d_in[pi++]; const int* i3_dst = (const int*)d_in[pi++];
  const int* batch_A  = (const int*)d_in[pi++];
  const int* batch_C2 = (const int*)d_in[pi++];
  const int* batch_C3 = (const int*)d_in[pi++];

  // workspace carve (floats)
  float* ws = (float*)d_ws;
  float* hA  = ws;                    // NA*256
  float* b0  = hA + NA * 256;         // NA*128
  float* b1  = b0 + NA * 128;         // NA*128
  float* b2  = b1 + NA * 128;         // NA*128
  float* el  = b2 + NA * 128;         // NA*4
  float* er  = el + NA * 4;           // NA*4
  float* mx  = er + NA * 4;           // NA*4
  float* dn  = mx + NA * 4;           // NA*4
  float* deg = dn + NA * 4;           // NA
  _Float16* Bp = (_Float16*)(deg + NA);  // up to (512/32)*8*32*16 = 65536 halves
  float* Xp  = deg + NA + 32768;      // NC2*384 padded-K buffer

  auto fill = [&](float* p, float v, long n) {
    k_fill<<<(unsigned)cdivl(n, 256), 256, 0, stream>>>(p, v, n);
  };
  // A:[M,Kp] f32 (rows stride Kp = K rounded to 32, zero-padded by caller if needed)
  auto gemm = [&](const float* A, const float* W, const float* bias, float* C,
                  int M, int N, int K, int ldc, int col0, int act) {
    int Kp = (K + 31) & ~31;
    int total = (Kp / 32) * (N / 16) * 32;
    k_prepB<<<(unsigned)cdivl(total, 256), 256, 0, stream>>>(W, Bp, K, Kp, N);
    unsigned g = (unsigned)cdivl(M, 128);
    if (N == 128)
      k_gemm<8><<<g, 256, 0, stream>>>(A, Bp, bias, C, M, Kp, ldc, col0, act);
    else
      k_gemm<4><<<g, 256, 0, stream>>>(A, Bp, bias, C, M, Kp, ldc, col0, act);
  };
  const float NEG = -INFINITY;

  // ================= bottom encoder (per edge type) =================
  for (int t = 0; t < 2; ++t) {
    const int* src = src_b[t]; const int* dst = dst_b[t];
    const float* hin = feats; int K = 64;
    for (int l = 0; l < 2; ++l) {
      gemm(hin, gat[t][l].W, nullptr, b1, (int)NA, 128, K, 128, 0, 0);  // z -> b1
      k_attn<<<(unsigned)cdivl(NA * 4, 256), 256, 0, stream>>>(b1, gat[t][l].al, gat[t][l].ar, el, er, NA * 4);
      fill(mx, NEG, NA * 4);
      k_edge_max<<<(unsigned)cdivl(EA * 4, 256), 256, 0, stream>>>(src, dst, el, er, mx, EA * 4);
      fill(dn, 0.f, NA * 4);
      float* numbuf = (l == 0) ? b0 : b2;
      fill(numbuf, 0.f, NA * 128);
      k_edge_soft<<<(unsigned)cdivl(EA * 4, 256), 256, 0, stream>>>(src, dst, el, er, mx, b1, dn, numbuf, EA * 4);
      k_gat_fin<<<(unsigned)cdivl(NA * 128, 256), 256, 0, stream>>>(dn, gat[t][l].b, numbuf, NA * 128);
      hin = numbuf; K = 128;
    }
    for (int l = 0; l < 2; ++l) {
      fill(b1, 0.f, NA * 128); fill(deg, 0.f, NA);
      k_scat_add4<<<(unsigned)cdivl(EA * 32, 256), 256, 0, stream>>>(src, dst, hin, b1, EA * 32, 32);
      k_deg<<<(unsigned)cdivl(EA, 256), 256, 0, stream>>>(dst, deg, EA);
      k_mean_div<<<(unsigned)cdivl(NA * 128, 256), 256, 0, stream>>>(b1, deg, NA * 128, 128);
      k_gin_pre<<<(unsigned)cdivl(NA * 128, 256), 256, 0, stream>>>(hin, b1, gin[t][l].eps, b1, NA * 128);
      if (l == 0) {
        float* obuf = (hin == b0) ? b2 : b0;
        gemm(b1, gin[t][l].W, gin[t][l].b, obuf, (int)NA, 128, 128, 128, 0, 1);
        hin = obuf;
      } else {
        gemm(b1, gin[t][l].W, gin[t][l].b, hA, (int)NA, 128, 128, 256, t * 128, 1);
      }
    }
  }

  // ================= A -> C2 max pool, concat pimg =================
  fill(b2, NEG, NC2 * 256);
  k_scat_max4<<<(unsigned)cdivl(EG1 * 64, 256), 256, 0, stream>>>(g1_src, g1_dst, hA, b2, EG1 * 64, 64);
  k_fix<<<(unsigned)cdivl(NC2 * 256, 256), 256, 0, stream>>>(b2, NC2 * 256);
  k_concat<<<(unsigned)cdivl(NC2 * 356, 256), 256, 0, stream>>>(b2, pimg, b0, NC2 * 356);

  // GIN 0 on I2 (K=356 -> pad A to 384)
  fill(b1, 0.f, NC2 * 356); fill(deg, 0.f, NC2);
  k_scat_add4<<<(unsigned)cdivl(EI2 * 89, 256), 256, 0, stream>>>(i2_src, i2_dst, b0, b1, EI2 * 89, 89);
  k_deg<<<(unsigned)cdivl(EI2, 256), 256, 0, stream>>>(i2_dst, deg, EI2);
  k_mean_div<<<(unsigned)cdivl(NC2 * 356, 256), 256, 0, stream>>>(b1, deg, NC2 * 356, 356);
  k_gin_pre<<<(unsigned)cdivl(NC2 * 356, 256), 256, 0, stream>>>(b0, b1, h2eps[0], b1, NC2 * 356);
  k_padcopy<<<(unsigned)cdivl(NC2 * 384, 256), 256, 0, stream>>>(b1, Xp, NC2, 356, 384);
  gemm(Xp, h2W[0], nullptr, b2, (int)NC2, 128, 356, 128, 0, 1);
  // GIN 1 on I2 (K=128)
  fill(b1, 0.f, NC2 * 128); fill(deg, 0.f, NC2);
  k_scat_add4<<<(unsigned)cdivl(EI2 * 32, 256), 256, 0, stream>>>(i2_src, i2_dst, b2, b1, EI2 * 32, 32);
  k_deg<<<(unsigned)cdivl(EI2, 256), 256, 0, stream>>>(i2_dst, deg, EI2);
  k_mean_div<<<(unsigned)cdivl(NC2 * 128, 256), 256, 0, stream>>>(b1, deg, NC2 * 128, 128);
  k_gin_pre<<<(unsigned)cdivl(NC2 * 128, 256), 256, 0, stream>>>(b2, b1, h2eps[1], b1, NC2 * 128);
  gemm(b1, h2W[1], nullptr, b0, (int)NC2, 128, 128, 128, 0, 1);   // final h2 -> b0

  // ================= C2 -> C3 max pool, GIN with BN =================
  float* c3h   = b1;
  float* c3agg = b1 + NC3 * 128;
  float* c3out = b1 + 2 * NC3 * 128;
  fill(c3h, NEG, NC3 * 128);
  k_scat_max4<<<(unsigned)cdivl(EG2 * 32, 256), 256, 0, stream>>>(g2_src, g2_dst, b0, c3h, EG2 * 32, 32);
  k_fix<<<(unsigned)cdivl(NC3 * 128, 256), 256, 0, stream>>>(c3h, NC3 * 128);
  fill(c3agg, 0.f, NC3 * 128); fill(deg, 0.f, NC3);
  k_scat_add4<<<(unsigned)cdivl(EI3 * 32, 256), 256, 0, stream>>>(i3_src, i3_dst, c3h, c3agg, EI3 * 32, 32);
  k_deg<<<(unsigned)cdivl(EI3, 256), 256, 0, stream>>>(i3_dst, deg, EI3);
  k_mean_div<<<(unsigned)cdivl(NC3 * 128, 256), 256, 0, stream>>>(c3agg, deg, NC3 * 128, 128);
  k_gin_pre<<<(unsigned)cdivl(NC3 * 128, 256), 256, 0, stream>>>(c3h, c3agg, h3eps, c3agg, NC3 * 128);
  gemm(c3agg, h3W0, nullptr, b2, (int)NC3, 128, 128, 128, 0, 2);  // lrelu
  k_bn_stats<<<1, 128, 0, stream>>>(b2, dn, (int)NC3, 128);       // dn reused for mu/var
  k_bn_apply<<<(unsigned)cdivl(NC3 * 128, 256), 256, 0, stream>>>(b2, dn, h3gamma, h3beta, NC3 * 128, 128);
  gemm(b2, h3W1, nullptr, c3out, (int)NC3, 128, 128, 128, 0, 1);  // final h3

  // ================= readouts + output MLP =================
  float* rbuf = el;        // 64*512
  float* tbuf = er;        // 64*128
  float* cA = mx; float* c2 = mx + 64; float* c3 = mx + 128;
  fill(rbuf, 0.f, 64 * 512);
  fill(cA, 0.f, 192);
  k_seg_sum4<<<(unsigned)cdivl(NA * 64, 256), 256, 0, stream>>>(batch_A, hA, rbuf, NA * 64, 64, 512, 0);
  k_seg_cnt<<<(unsigned)cdivl(NA, 256), 256, 0, stream>>>(batch_A, cA, NA);
  k_seg_sum4<<<(unsigned)cdivl(NC2 * 32, 256), 256, 0, stream>>>(batch_C2, b0, rbuf, NC2 * 32, 32, 512, 256);
  k_seg_cnt<<<(unsigned)cdivl(NC2, 256), 256, 0, stream>>>(batch_C2, c2, NC2);
  k_seg_sum4<<<(unsigned)cdivl(NC3 * 32, 256), 256, 0, stream>>>(batch_C3, c3out, rbuf, NC3 * 32, 32, 512, 384);
  k_seg_cnt<<<(unsigned)cdivl(NC3, 256), 256, 0, stream>>>(batch_C3, c3, NC3);
  k_rdiv<<<(unsigned)cdivl(64 * 512, 256), 256, 0, stream>>>(rbuf, cA, c2, c3, 64 * 512);

  gemm(rbuf, oW0, nullptr, tbuf, 64, 128, 512, 128, 0, 2);        // lrelu
  gemm(tbuf, oW1, nullptr, (float*)d_out, 64, 64, 128, 64, 0, 0); // [64,64]
}
